// CoordVarAwareStepPredictor_1254130450675
// MI455X (gfx1250) — matrix operand order repaired and checked by reading, not compiled, not run on hardware
//
#include <hip/hip_runtime.h>
#include <math.h>
#include <stdint.h>

#define NC   4096      // n_coord (rows of X, K dim of GEMM)
#define NP   16384     // trajectory points (cols of X, N dim of GEMM)
#define HID  64        // hidden
#define NTOP 2048
#define BN   128       // columns per block in GEMM
#define BK   32        // K step (WMMA bf16 K)

typedef __attribute__((ext_vector_type(16))) __bf16 v16bf;
typedef __attribute__((ext_vector_type(8)))  __bf16 bf16x8;
typedef __attribute__((ext_vector_type(8)))  float  v8f;
typedef __attribute__((ext_vector_type(4)))  float  f32x4;
typedef __attribute__((ext_vector_type(4)))  unsigned int u32x4;
typedef __attribute__((ext_vector_type(8)))  int    i32x8;
typedef __attribute__((ext_vector_type(4)))  int    i32x4;

#if __has_builtin(__builtin_amdgcn_tensor_load_to_lds)
#define USE_TDM 1
#else
#define USE_TDM 0
#endif

// ---------------- init: zero accumulators (must happen every call) ----------
__global__ void init_kernel(float* S, float* hsum) {
    int t = threadIdx.x;
    if (t == 0) S[0] = 0.0f;
    if (t < HID) hsum[t] = 0.0f;
}

// ---------------- per-row variance (ddof=1) --------------------------------
__global__ __launch_bounds__(256) void rowvar_kernel(const float* __restrict__ X,
                                                     float* __restrict__ var) {
    int n = blockIdx.x;
    int t = threadIdx.x;
    float s = 0.0f, q = 0.0f;
    const float* row = X + (size_t)n * NP;
#pragma unroll 4
    for (int r = 0; r < NP / 256; ++r) {
        float x = row[t + 256 * r];
        s += x; q += x * x;
    }
    __shared__ float s1[256], s2[256];
    s1[t] = s; s2[t] = q;
    __syncthreads();
    for (int off = 128; off > 0; off >>= 1) {
        if (t < off) { s1[t] += s1[t + off]; s2[t] += s2[t + off]; }
        __syncthreads();
    }
    if (t == 0) {
        float sum = s1[0], sumsq = s2[0];
        var[n] = (sumsq - sum * sum * (1.0f / NP)) * (1.0f / (NP - 1));
    }
}

// ---------------- per-column L2 norms, summed into S ------------------------
__global__ __launch_bounds__(256) void colnorm_kernel(const float* __restrict__ X,
                                                      float* __restrict__ S) {
    int t = threadIdx.x;
    int p = blockIdx.x * 256 + t;
    float q = 0.0f;
#pragma unroll 4
    for (int n = 0; n < NC; ++n) {
        float x = X[(size_t)n * NP + p];
        q += x * x;
    }
    float v = sqrtf(q);
    __shared__ float s1[256];
    s1[t] = v;
    __syncthreads();
    for (int off = 128; off > 0; off >>= 1) {
        if (t < off) s1[t] += s1[t + off];
        __syncthreads();
    }
    if (t == 0) atomicAdd(S, s1[0]);
}

// ---------------- bitonic argsort of 4096 keys (ascending on -var) ----------
__global__ __launch_bounds__(1024) void sort_kernel(const float* __restrict__ var,
                                                    int* __restrict__ uid) {
    __shared__ float key[NC];
    __shared__ int   idx[NC];
    int t = threadIdx.x;
    for (int i = t; i < NC; i += 1024) { key[i] = -var[i]; idx[i] = i; }
    __syncthreads();
    for (int k = 2; k <= NC; k <<= 1) {
        for (int j = k >> 1; j > 0; j >>= 1) {
            for (int i = t; i < NC; i += 1024) {
                int l = i ^ j;
                if (l > i) {
                    bool up = ((i & k) == 0);
                    float ki = key[i], kl = key[l];
                    if ((ki > kl) == up) {
                        key[i] = kl; key[l] = ki;
                        int tmp = idx[i]; idx[i] = idx[l]; idx[l] = tmp;
                    }
                }
            }
            __syncthreads();
        }
    }
    for (int i = t; i < NC; i += 1024) uid[i] = idx[i];
}

// ---------------- perm_inv: perm[i] = uid[(i+NTOP)&(NC-1)] ------------------
__global__ void perminv_kernel(const int* __restrict__ uid, int* __restrict__ pinv) {
    int i = blockIdx.x * blockDim.x + threadIdx.x;
    if (i < NC) pinv[uid[(i + NTOP) & (NC - 1)]] = i;
}

// ---------------- W1p[h][q] = bf16(W1[h][pinv[q]]) --------------------------
__global__ void buildw1p_kernel(const float* __restrict__ W1,
                                const int* __restrict__ pinv,
                                __bf16* __restrict__ W1p) {
    int e = blockIdx.x * blockDim.x + threadIdx.x;   // 64*4096 elements
    if (e < HID * NC) {
        int h = e >> 12, q = e & (NC - 1);
        W1p[e] = (__bf16)W1[h * (NC + 1) + pinv[q]];
    }
}

// ---------------- fused GEMM + tanh + column reduction ----------------------
// C[h,p] = sum_q W1p[h,q] * X[q,p]  (bf16 WMMA, f32 acc)
// hsum[h] += sum_p tanh(C[h,p]/S + losses[p]*W1[h,NC] + b1[h])
__global__ __launch_bounds__(256) void gemm_kernel(const float* __restrict__ X,
                                                   const float* __restrict__ losses,
                                                   const float* __restrict__ W1,
                                                   const float* __restrict__ b1,
                                                   const __bf16* __restrict__ W1p,
                                                   const float* __restrict__ S,
                                                   float* __restrict__ hsum) {
    __shared__ __bf16 As[HID][BK];     // A tile, row-major [64][32]
    __shared__ __bf16 Bt[BN][BK];      // B tile transposed: Bt[c][kk] = X[k0+kk, col]
    int t = threadIdx.x;
    int lane = t & 31;
    int w = t >> 5;                              // wave 0..7
    int colbase = blockIdx.x * BN;
    int half = lane >> 4;                        // 0: lanes 0-15, 1: lanes 16-31

    v8f acc[4];
#pragma unroll
    for (int m = 0; m < 4; ++m) acc[m] = (v8f){0.f,0.f,0.f,0.f,0.f,0.f,0.f,0.f};

#if USE_TDM
    // ---- Tensor DMA descriptor for the A tile (2D: 4096x64 bf16, tile 32x64) ----
    unsigned int ldsA = (unsigned int)(uintptr_t)(&As[0][0]);
    unsigned long long gbase = (unsigned long long)(uintptr_t)W1p;
    i32x8 g1;
    g1[0] = 0x00010000;                 // workgroup_mask=0, data_size=1 (2 bytes)
    g1[1] = (int)(4096u << 16);         // tensor_dim0[15:0]=4096 in bits[63:48]
    g1[2] = (int)(64u << 16);           // tensor_dim1[15:0]=64 in bits[95:80]
    g1[3] = (int)(32u << 16);           // tile_dim0=32 in bits[127:112]
    g1[4] = 64;                         // tile_dim1=64 in bits[143:128]
    g1[5] = 4096;                       // tensor_dim0_stride[31:0]=4096
    g1[6] = 0;
    g1[7] = 0;
    i32x4 gz = {0, 0, 0, 0};            // groups 2/3 unused (2D tensor)
#endif

    for (int k0 = 0; k0 < NC; k0 += BK) {
        __syncthreads();
#if USE_TDM
        if (w == 0) {                   // one TDM op per workgroup per K-step
            unsigned long long ga = gbase + (unsigned long long)(2 * k0);
            u32x4 g0;
            g0[0] = 1u;                                        // count=1, load, not restore
            g0[1] = ldsA;                                      // lds_addr
            g0[2] = (unsigned int)ga;                          // global_addr[31:0]
            g0[3] = (unsigned int)((ga >> 32) & 0x01FFFFFFu) | (2u << 30);  // addr hi + type=2
#if __has_include(<hip/amd_detail/amd_gfx1250_TDM.h>)
            i32x8 gz8 = {0, 0, 0, 0, 0, 0, 0, 0};
            __builtin_amdgcn_tensor_load_to_lds(g0, g1, gz, gz, gz8, 0);
#else
            __builtin_amdgcn_tensor_load_to_lds(g0, g1, gz, gz, 0);
#endif
        }
#else
        {   // vectorized fallback: 16B per thread (h = t/4, kk = (t%4)*8)
            int h = t >> 2, kk = (t & 3) * 8;
            *(u32x4*)&As[h][kk] = *(const u32x4*)(W1p + (size_t)h * NC + k0 + kk);
        }
#endif
        // --- stage B: f32 -> bf16, transposed into LDS, vectorized ---
        // group g covers a (2 rows x 4 cols) patch: 2x global_load_b128 + 4x ds_store_b32
#pragma unroll
        for (int r = 0; r < 2; ++r) {
            int g = t + 256 * r;                 // 512 groups
            int kkp = g >> 5;                    // 0..15 (row pair)
            int c4 = (g & 31) * 4;               // 0..124
            const float* rowp = X + (size_t)(k0 + 2 * kkp) * NP + colbase + c4;
            f32x4 a = *(const f32x4*)rowp;
            f32x4 b = *(const f32x4*)(rowp + NP);
#pragma unroll
            for (int j = 0; j < 4; ++j) {
                unsigned int lo = __builtin_bit_cast(unsigned short, (__bf16)a[j]);
                unsigned int hi = __builtin_bit_cast(unsigned short, (__bf16)b[j]);
                *(unsigned int*)&Bt[c4 + j][2 * kkp] = (hi << 16) | lo;
            }
        }
        if (k0 + BK < NC)
            __builtin_prefetch(&X[(size_t)(k0 + BK + (t >> 7)) * NP + colbase + (t & 127)], 0, 1);
#if USE_TDM
        __builtin_amdgcn_s_wait_tensorcnt(0);
#endif
        __syncthreads();

        // --- fragments + WMMA ---
        int c = w * 16 + (lane & 15);
        v16bf bfrag = *(const v16bf*)&Bt[c][16 * half];   // 16 K-contiguous per lane
#pragma unroll
        for (int m = 0; m < 4; ++m) {
            int h = m * 16 + (lane & 15);
            bf16x8 lo = *(const bf16x8*)&As[h][16 * half];
            bf16x8 hi = *(const bf16x8*)&As[h][16 * half + 8];
            v16bf afrag;
#pragma unroll
            for (int i = 0; i < 8; ++i) { afrag[i] = lo[i]; afrag[8 + i] = hi[i]; }
            acc[m] = __builtin_amdgcn_wmma_f32_16x16x32_bf16(
                false, afrag, false, bfrag, (short)0, acc[m], false, false);
        }
    }

    // --- epilogue: tanh, reduce over the 16 columns, accumulate hsum ---
    float invS = 1.0f / S[0];
    int p = colbase + w * 16 + (lane & 15);
    float lp = losses[p];
#pragma unroll
    for (int m = 0; m < 4; ++m) {
#pragma unroll
        for (int i = 0; i < 8; ++i) {
            int h = m * 16 + i + 8 * half;
            float v = tanhf(acc[m][i] * invS + lp * W1[h * (NC + 1) + NC] + b1[h]);
            v += __shfl_xor(v, 1);
            v += __shfl_xor(v, 2);
            v += __shfl_xor(v, 4);
            v += __shfl_xor(v, 8);
            if ((lane & 15) == 0) atomicAdd(&hsum[h], v);
        }
    }
}

// ---------------- finalize: W2 projection + scatter + *S --------------------
__global__ void finalize_kernel(const float* __restrict__ hsum,
                                const float* __restrict__ W2,
                                const float* __restrict__ b2,
                                const int* __restrict__ uid,
                                const float* __restrict__ S,
                                float* __restrict__ out) {
    int i = blockIdx.x * blockDim.x + threadIdx.x;
    if (i < NC) {
        float u = b2[i];
#pragma unroll
        for (int h = 0; h < HID; ++h) u += hsum[h] * W2[i * HID + h];
        out[uid[i]] = u * S[0];
    }
}

extern "C" void kernel_launch(void* const* d_in, const int* in_sizes, int n_in,
                              void* d_out, int out_size, void* d_ws, size_t ws_size,
                              hipStream_t stream) {
    const float* X      = (const float*)d_in[0];   // points_rel [4096,16384]
    const float* losses = (const float*)d_in[1];   // [16384]
    const float* W1     = (const float*)d_in[2];   // [64,4097]
    const float* b1     = (const float*)d_in[3];   // [64]
    const float* W2     = (const float*)d_in[4];   // [4096,64]
    const float* b2     = (const float*)d_in[5];   // [4096]
    float* out = (float*)d_out;

    char* ws = (char*)d_ws;
    float*  d_var  = (float*)(ws + 0);             // 16 KB
    int*    d_uid  = (int*)(ws + 16384);           // 16 KB
    int*    d_pinv = (int*)(ws + 32768);           // 16 KB
    float*  d_S    = (float*)(ws + 49152);         // 4 B
    float*  d_hsum = (float*)(ws + 49408);         // 256 B
    __bf16* d_W1p  = (__bf16*)(ws + 65536);        // 512 KB

    init_kernel<<<1, 128, 0, stream>>>(d_S, d_hsum);
    rowvar_kernel<<<NC, 256, 0, stream>>>(X, d_var);
    colnorm_kernel<<<NP / 256, 256, 0, stream>>>(X, d_S);
    sort_kernel<<<1, 1024, 0, stream>>>(d_var, d_uid);
    perminv_kernel<<<NC / 256, 256, 0, stream>>>(d_uid, d_pinv);
    buildw1p_kernel<<<(HID * NC) / 256, 256, 0, stream>>>(W1, d_pinv, d_W1p);
    gemm_kernel<<<NP / BN, 256, 0, stream>>>(X, losses, W1, b1, d_W1p, d_S, d_hsum);
    finalize_kernel<<<NC / 256, 256, 0, stream>>>(d_hsum, W2, b2, d_uid, d_S, out);
}
